// MLA_33827162423865
// MI455X (gfx1250) — matrix operand order repaired and checked by reading, not compiled, not run on hardware
//
#include <hip/hip_runtime.h>
#include <hip/hip_bf16.h>
#include <math.h>

// ---- problem constants (from reference) ----
#define B_     8
#define DIM_   2048
#define H_     16
#define QLR_   1536
#define KVLR_  512
#define DN_    128
#define DR_    64
#define DV_    128
#define START_ 8191
#define T_     8192          // START + 1 new token
#define EPS_   1e-6f
#define SCALE_ 0.07216878364870322f   // (DN+DR)^-0.5 = 1/sqrt(192)
#define HD_    (DN_ + DR_)   // 192 per-head q dim
#define KW_    (DN_ + DV_)   // 256 rows per head in wkv_b

#define TBLK_  512           // context kernel: T-range per block
#define NSPL_  (T_ / TBLK_)  // 16 z-splits
#define NPART_ (NSPL_ * 2)   // 32 T-partials (2 per block: wave t-halves)
#define SROW_  516           // LDS row stride (floats): 16B aligned, 4-bank skew

typedef __attribute__((ext_vector_type(16))) __bf16 v16bf;
typedef __attribute__((ext_vector_type(8)))  float  v8f;

__device__ __forceinline__ __bf16 tobf(float f) { return (__bf16)f; }

// CDNA5 async global->LDS copy of 16 bytes per lane (GV mode), ASYNCcnt-tracked.
__device__ __forceinline__ void async_copy_b128(uint32_t lds_byte_addr, const void* gsrc) {
  asm volatile("global_load_async_to_lds_b128 %0, %1, off"
               :: "v"(lds_byte_addr), "v"(gsrc) : "memory");
}
__device__ __forceinline__ void wait_async0() {
  asm volatile("s_wait_asynccnt 0x0" ::: "memory");
}

// ---------------------------------------------------------------------------
// Generic GEMV: out[b][row] = dot(W[row,:K], xin[b,:K]) + bias[row]
// wave-per-row, float4 loads, shuffle reduction. grid = (N/8, B), block = 256.
// ---------------------------------------------------------------------------
__global__ void gemv_kernel(const float* __restrict__ W, const float* __restrict__ xin,
                            const float* __restrict__ bias, float* __restrict__ out,
                            int N, int K) {
  int b = blockIdx.y;
  int w = threadIdx.x >> 5, lane = threadIdx.x & 31;
  int row = blockIdx.x * 8 + w;
  if (row >= N) return;
  const float4* Wr = (const float4*)(W + (size_t)row * K);
  const float4* xr = (const float4*)(xin + (size_t)b * K);
  float acc = 0.f;
  int K4 = K >> 2;
  for (int k = lane; k < K4; k += 32) {
    float4 a = Wr[k], x4 = xr[k];
    acc += a.x * x4.x + a.y * x4.y + a.z * x4.z + a.w * x4.w;
  }
  for (int off = 16; off > 0; off >>= 1) acc += __shfl_down(acc, off, 32);
  if (lane == 0) out[(size_t)b * N + row] = acc + bias[row];
}

// ---------------------------------------------------------------------------
// RMS norm: out[b][i] = in[b][i] * w[i] * rsqrt(mean(in[b]^2) + eps)
// ---------------------------------------------------------------------------
__global__ void rms_kernel(const float* __restrict__ in, const float* __restrict__ w,
                           float* __restrict__ out, int len, int in_stride, int out_stride) {
  __shared__ float red[256];
  int b = blockIdx.x, tid = threadIdx.x;
  const float* xi = in + (size_t)b * in_stride;
  float s = 0.f;
  for (int i = tid; i < len; i += 256) { float v = xi[i]; s += v * v; }
  red[tid] = s; __syncthreads();
  for (int off = 128; off > 0; off >>= 1) {
    if (tid < off) red[tid] += red[tid + off];
    __syncthreads();
  }
  float r = rsqrtf(red[0] / (float)len + EPS_);
  for (int i = tid; i < len; i += 256)
    out[(size_t)b * out_stride + i] = xi[i] * w[i] * r;
}

// ---------------------------------------------------------------------------
// RoPE (S=1): in-place on q_pe (per head), kvpe tail -> pe_new.
// ---------------------------------------------------------------------------
__global__ void rope_kernel(float* __restrict__ q, const float* __restrict__ kvpe,
                            float* __restrict__ pe_new,
                            const float* __restrict__ fc, const float* __restrict__ fs) {
  int b = blockIdx.x, t = threadIdx.x;
  if (t < H_ * 32) {
    int h = t >> 5, j = t & 31;
    float* qp = q + (size_t)b * (H_ * HD_) + h * HD_ + DN_;
    float xr = qp[2 * j], xi = qp[2 * j + 1], c = fc[j], s = fs[j];
    qp[2 * j]     = xr * c - xi * s;
    qp[2 * j + 1] = xr * s + xi * c;
  } else if (t < H_ * 32 + 32) {
    int j = t - H_ * 32;
    const float* kp = kvpe + (size_t)b * (KVLR_ + DR_) + KVLR_;
    float xr = kp[2 * j], xi = kp[2 * j + 1], c = fc[j], s = fs[j];
    pe_new[(size_t)b * DR_ + 2 * j]     = xr * c - xi * s;
    pe_new[(size_t)b * DR_ + 2 * j + 1] = xr * s + xi * c;
  }
}

// ---------------------------------------------------------------------------
// q_abs[b,h,c] = sum_d q_nope[b,h,d] * wkv_b[h,d,c]   grid=(B,H), block=256
// ---------------------------------------------------------------------------
__global__ void qabs_kernel(const float* __restrict__ q, const float* __restrict__ wkvb,
                            float* __restrict__ qabs) {
  __shared__ float qn[DN_];
  int b = blockIdx.x, h = blockIdx.y, tid = threadIdx.x;
  if (tid < DN_) qn[tid] = q[(size_t)b * (H_ * HD_) + h * HD_ + tid];
  __syncthreads();
  for (int c = tid; c < KVLR_; c += 256) {
    float acc = 0.f;
    for (int d = 0; d < DN_; ++d)
      acc += qn[d] * wkvb[((size_t)h * KW_ + d) * KVLR_ + c];
    qabs[(((size_t)b * H_) + h) * KVLR_ + c] = acc;
  }
}

// ---------------------------------------------------------------------------
// Scores: per b, [16h x 576] x [576 x T] GEMM with bf16 WMMA, fp32 accum.
// grid = (T/128, B), block = 256 (8 waves, one 16x16 t-tile per wave).
// Per-lane cache-row pointers hoisted; 16 kv chunks + 2 unrolled pe chunks.
// ---------------------------------------------------------------------------
__global__ void __launch_bounds__(256)
scores_kernel(const float* __restrict__ qabs, const float* __restrict__ qbuf,
              const float* __restrict__ kvpre, const float* __restrict__ pepre,
              const float* __restrict__ kvnew, const float* __restrict__ penew,
              float* __restrict__ scores) {
  int b = blockIdx.y;
  int w = threadIdx.x >> 5, lane = threadIdx.x & 31;
  int half = lane >> 4, mn = lane & 15;
  int t0 = blockIdx.x * 128 + w * 16;
  int t = t0 + mn;                         // B-operand column -> time index
  v8f acc = {0.f, 0.f, 0.f, 0.f, 0.f, 0.f, 0.f, 0.f};

  // loop-invariant row pointers (per lane)
  const float* arow_kv = qabs + (((size_t)b * H_) + mn) * KVLR_;          // A: m = head
  const float* arow_pe = qbuf + (size_t)b * (H_ * HD_) + mn * HD_ + DN_;
  const float* bkv = (t < START_) ? kvpre + ((size_t)b * START_ + t) * KVLR_
                                  : kvnew + (size_t)b * KVLR_;
  const float* bpe = (t < START_) ? pepre + ((size_t)b * START_ + t) * DR_
                                  : penew + (size_t)b * DR_;

  // ---- 16 chunks over the 512-dim kv latent ----
  for (int kk = 0; kk < 16; ++kk) {
    int kbase = kk * 32;
    v16bf a;
#pragma unroll
    for (int v = 0; v < 8; ++v) {
      int k = kbase + half * 16 + v * 2;
      a[2 * v]     = tobf(arow_kv[k]);
      a[2 * v + 1] = tobf(arow_kv[k + 1]);
    }
    int k0 = kbase + half * 16;
    __builtin_prefetch(bkv + k0 + 128, 0, 0);   // stream-ahead (global_prefetch_b8)
    v16bf bf;
#pragma unroll
    for (int j = 0; j < 16; ++j) bf[j] = tobf(bkv[k0 + j]);
    acc = __builtin_amdgcn_wmma_f32_16x16x32_bf16(false, a, false, bf,
                                                  (short)0, acc, false, false);
  }
  // ---- 2 chunks over the 64-dim rope part ----
#pragma unroll
  for (int kk = 0; kk < 2; ++kk) {
    int kbase = kk * 32;
    v16bf a;
#pragma unroll
    for (int v = 0; v < 8; ++v) {
      int k = kbase + half * 16 + v * 2;
      a[2 * v]     = tobf(arow_pe[k]);
      a[2 * v + 1] = tobf(arow_pe[k + 1]);
    }
    int k0 = kbase + half * 16;
    v16bf bf;
#pragma unroll
    for (int j = 0; j < 16; ++j) bf[j] = tobf(bpe[k0 + j]);
    acc = __builtin_amdgcn_wmma_f32_16x16x32_bf16(false, a, false, bf,
                                                  (short)0, acc, false, false);
  }
#pragma unroll
  for (int g = 0; g < 8; ++g) {
    int h = half * 8 + g;              // D layout: M = g + half*8, N = lane&15
    scores[(((size_t)b * H_) + h) * T_ + t0 + mn] = acc[g] * SCALE_;
  }
}

// ---------------------------------------------------------------------------
// Softmax over T per (b,h). grid = B*H, block = 256. In-place.
// ---------------------------------------------------------------------------
__global__ void softmax_kernel(float* __restrict__ s) {
  __shared__ float red[256];
  int tid = threadIdx.x;
  float* row = s + (size_t)blockIdx.x * T_;
  float mx = -1e30f;
  for (int i = tid; i < T_; i += 256) mx = fmaxf(mx, row[i]);
  red[tid] = mx; __syncthreads();
  for (int off = 128; off > 0; off >>= 1) {
    if (tid < off) red[tid] = fmaxf(red[tid], red[tid + off]);
    __syncthreads();
  }
  mx = red[0]; __syncthreads();
  float sum = 0.f;
  for (int i = tid; i < T_; i += 256) { float e = __expf(row[i] - mx); row[i] = e; sum += e; }
  red[tid] = sum; __syncthreads();
  for (int off = 128; off > 0; off >>= 1) {
    if (tid < off) red[tid] += red[tid + off];
    __syncthreads();
  }
  float inv = 1.f / red[0];
  for (int i = tid; i < T_; i += 256) row[i] *= inv;
}

// ---------------------------------------------------------------------------
// Context: per b, [16h x T] x [T x 512c] with bf16 WMMA, fp32 accum.
// grid = (8 c-tiles of 64, B, 16 z-splits of 512 t), block = 256.
// attn slice for the block's T-range is async-staged into LDS once
// (global_load_async_to_lds_b128), then A-fragments come from LDS.
// wave w: n_sub = w&3 (c sub-tile), t_half = w>>2 (256 t's -> 8 chunks of 32).
// 32 deterministic T-partials, summed later (no atomics).
// ---------------------------------------------------------------------------
__global__ void __launch_bounds__(256)
context_kernel(const float* __restrict__ attn, const float* __restrict__ kvpre,
               const float* __restrict__ kvnew, float* __restrict__ ctx_partial) {
  __shared__ float sattn[H_ * SROW_];        // 16 rows x 516 floats = 33 KB
  int b = blockIdx.y;
  int tid = threadIdx.x;
  int w = tid >> 5, lane = tid & 31;
  int half = lane >> 4, mn = lane & 15;
  int n_sub = w & 3, t_half = w >> 2;
  int c0 = blockIdx.x * 64 + n_sub * 16;
  int tblock = blockIdx.z * TBLK_;
  int p = blockIdx.z * 2 + t_half;
  int c = c0 + mn;                           // B-operand column

  // ---- async-stage attn[b, :, tblock..tblock+511] into LDS ----
  {
    const float* abase = attn + ((size_t)b * H_) * T_ + tblock;
    for (int i = tid; i < H_ * (TBLK_ / 4); i += 256) {   // 16-byte chunks
      int h = i >> 7, cc = i & 127;                        // 128 chunks per row
      const float* g = abase + (size_t)h * T_ + cc * 4;
      uint32_t l = (uint32_t)(uintptr_t)(&sattn[h * SROW_ + cc * 4]);
      async_copy_b128(l, g);
    }
    wait_async0();
    __syncthreads();
  }

  v8f acc = {0.f, 0.f, 0.f, 0.f, 0.f, 0.f, 0.f, 0.f};
  const float* arow = &sattn[mn * SROW_ + t_half * (TBLK_ / 2)];   // A: m = head

  for (int it = 0; it < 8; ++it) {
    int tc = it * 32;                                    // local within wave range
    v16bf a;
#pragma unroll
    for (int v = 0; v < 8; ++v) {
      int k = tc + half * 16 + v * 2;
      a[2 * v]     = tobf(arow[k]);
      a[2 * v + 1] = tobf(arow[k + 1]);
    }
    v16bf bf;
    int k0 = tblock + t_half * (TBLK_ / 2) + tc + half * 16;   // global time index
    if (k0 + 15 < START_) {                  // fast path: all rows in prefix
      const float* br = kvpre + ((size_t)b * START_ + k0) * KVLR_ + c;
      __builtin_prefetch(br + (size_t)32 * KVLR_, 0, 0);
#pragma unroll
      for (int j = 0; j < 16; ++j) bf[j] = tobf(br[(size_t)j * KVLR_]);
    } else {                                 // tail: includes new token row
#pragma unroll
      for (int j = 0; j < 16; ++j) {
        int tt = k0 + j;
        bf[j] = tobf((tt < START_) ? kvpre[((size_t)b * START_ + tt) * KVLR_ + c]
                                   : kvnew[(size_t)b * KVLR_ + c]);
      }
    }
    acc = __builtin_amdgcn_wmma_f32_16x16x32_bf16(false, a, false, bf,
                                                  (short)0, acc, false, false);
  }
#pragma unroll
  for (int g = 0; g < 8; ++g) {
    int h = half * 8 + g;
    ctx_partial[((((size_t)p * B_) + b) * H_ + h) * KVLR_ + c] = acc[g];
  }
}

// ---------------------------------------------------------------------------
// o[b, h*DV+d] = sum_c (sum_p ctx_partial[p,b,h,c]) * wkv_b[h, DN+d, c]
// grid = (B, H), block = 128
// ---------------------------------------------------------------------------
__global__ void ovalue_kernel(const float* __restrict__ ctx_partial,
                              const float* __restrict__ wkvb, float* __restrict__ o) {
  __shared__ float ctx[KVLR_];
  int b = blockIdx.x, h = blockIdx.y, tid = threadIdx.x;
  for (int cc = tid; cc < KVLR_; cc += 128) {
    float s = 0.f;
    for (int p = 0; p < NPART_; ++p)
      s += ctx_partial[((((size_t)p * B_) + b) * H_ + h) * KVLR_ + cc];
    ctx[cc] = s;
  }
  __syncthreads();
  int d = tid;
  float acc = 0.f;
  const float* wr = wkvb + ((size_t)h * KW_ + DN_ + d) * KVLR_;
  for (int cc = 0; cc < KVLR_; ++cc) acc += ctx[cc] * wr[cc];
  o[(size_t)b * (H_ * DV_) + h * DV_ + d] = acc;
}

// ---------------------------------------------------------------------------
extern "C" void kernel_launch(void* const* d_in, const int* in_sizes, int n_in,
                              void* d_out, int out_size, void* d_ws, size_t ws_size,
                              hipStream_t stream) {
  const float* x      = (const float*)d_in[0];
  const float* fcos   = (const float*)d_in[2];
  const float* fsin   = (const float*)d_in[3];
  const float* kvpre  = (const float*)d_in[4];
  const float* pepre  = (const float*)d_in[5];
  const float* wqa    = (const float*)d_in[6];
  const float* wqa_b  = (const float*)d_in[7];
  const float* qnw    = (const float*)d_in[8];
  const float* wqb    = (const float*)d_in[9];
  const float* wqb_b  = (const float*)d_in[10];
  const float* wkva   = (const float*)d_in[11];
  const float* wkva_b = (const float*)d_in[12];
  const float* kvnw   = (const float*)d_in[13];
  const float* wkvb   = (const float*)d_in[14];
  const float* wo     = (const float*)d_in[15];
  const float* wo_b   = (const float*)d_in[16];
  float* out = (float*)d_out;

  // workspace layout (floats), all offsets 16-float aligned
  float* ws  = (float*)d_ws;
  float* qa       = ws;                        // B*1536
  float* q        = qa + B_ * QLR_;            // B*3072
  float* qabs     = q + B_ * H_ * HD_;         // B*16*512
  float* kvpe     = qabs + B_ * H_ * KVLR_;    // B*576
  float* kvnew    = kvpe + B_ * (KVLR_ + DR_); // B*512
  float* penew    = kvnew + B_ * KVLR_;        // B*64
  float* scores   = penew + B_ * DR_;          // B*16*8192 (becomes attn)
  float* ctx_part = scores + (size_t)B_ * H_ * T_;             // 32*B*16*512
  float* oflat    = ctx_part + (size_t)NPART_ * B_ * H_ * KVLR_; // B*2048

  // 1. q latent + kv latent projections
  gemv_kernel<<<dim3(QLR_ / 8, B_), 256, 0, stream>>>(wqa, x, wqa_b, qa, QLR_, DIM_);
  gemv_kernel<<<dim3((KVLR_ + DR_) / 8, B_), 256, 0, stream>>>(wkva, x, wkva_b, kvpe,
                                                               KVLR_ + DR_, DIM_);
  // 2. RMS norms
  rms_kernel<<<B_, 256, 0, stream>>>(qa, qnw, qa, QLR_, QLR_, QLR_);
  rms_kernel<<<B_, 256, 0, stream>>>(kvpe, kvnw, kvnew, KVLR_, KVLR_ + DR_, KVLR_);
  // 3. q up-projection
  gemv_kernel<<<dim3((H_ * HD_) / 8, B_), 256, 0, stream>>>(wqb, qa, wqb_b, q,
                                                            H_ * HD_, QLR_);
  // 4. RoPE (q_pe in place, k_pe -> penew)
  rope_kernel<<<B_, 544, 0, stream>>>(q, kvpe, penew, fcos, fsin);
  // 5. absorb q_nope through wkv_b nope block
  qabs_kernel<<<dim3(B_, H_), 256, 0, stream>>>(q, wkvb, qabs);
  // 6. attention scores (WMMA)
  scores_kernel<<<dim3(T_ / 128, B_), 256, 0, stream>>>(qabs, q, kvpre, pepre,
                                                        kvnew, penew, scores);
  // 7. softmax
  softmax_kernel<<<B_ * H_, 256, 0, stream>>>(scores);
  // 8. context (WMMA + async LDS staging, 32 deterministic T-partials)
  context_kernel<<<dim3(KVLR_ / 64, B_, NSPL_), 256, 0, stream>>>(scores, kvpre,
                                                                  kvnew, ctx_part);
  // 9. value absorb
  ovalue_kernel<<<dim3(B_, H_), 128, 0, stream>>>(ctx_part, wkvb, oflat);
  // 10. output projection
  gemv_kernel<<<dim3(DIM_ / 8, B_), 256, 0, stream>>>(wo, oflat, wo_b, out,
                                                      DIM_, DIM_);
}
